// CEC_10101763080503
// MI455X (gfx1250) — matrix operand order repaired and checked by reading, not compiled, not run on hardware
//
#include <hip/hip_runtime.h>

// CondConv on MI455X (gfx1250):
//   pool -> routing(sigmoid) -> per-sample bf16 kernels (K-permuted) -> implicit-GEMM conv
//   conv GEMM per sample: M=256 (Cout), N=3136 (56x56), K=1152, K-order k' = (r*3+s)*128 + c
//   v_wmma_f32_16x16x32_bf16, wave32, 8 waves/block, BM=256 BN=64 BK=32.
//   A tile: global_load_async_to_lds_b128, 2-deep prefetch (3 LDS buffers,
//   s_wait_asynccnt 4 slack, in-order async completion per ISA 08 §4.1).
//   LDS rows padded to 40 bf16 (80 B) -> conflict-free 64-bank access.
//   Fragment LDS reads issued as explicit uint4 -> guaranteed ds_load_b128.

typedef __bf16 bhalf;
typedef __attribute__((ext_vector_type(16))) __bf16 bf16x16;
typedef __attribute__((ext_vector_type(8)))  float  f32x8;

#define AS3 __attribute__((address_space(3)))

#define B_      32
#define CIN     128
#define HW_IN   112
#define PIX_IN  (HW_IN * HW_IN)       // 12544
#define COUT    256
#define E_      4
#define HW_OUT  56
#define PIX_OUT (HW_OUT * HW_OUT)     // 3136
#define KDIM    (CIN * 9)             // 1152
#define WPS     (COUT * KDIM)         // 294912
#define KSTEPS  (KDIM / 32)           // 36
#define LDSTR   40                    // padded LDS row stride (bf16 elements)

// ---------------------------------------------------------------- pooling ---
__global__ __launch_bounds__(256) void CEC_pool_kernel(const float* __restrict__ x,
                                                       float* __restrict__ pooled) {
    const int bc = blockIdx.x;                    // b*128 + c
    const float* p = x + (size_t)bc * PIX_IN;
    float s = 0.0f;
    for (int i = threadIdx.x; i < PIX_IN; i += 256) s += p[i];
    __shared__ float red[8];
    #pragma unroll
    for (int off = 16; off > 0; off >>= 1) s += __shfl_down(s, off);
    if ((threadIdx.x & 31) == 0) red[threadIdx.x >> 5] = s;
    __syncthreads();
    if (threadIdx.x < 8) {
        s = red[threadIdx.x];
        #pragma unroll
        for (int off = 4; off > 0; off >>= 1) s += __shfl_down(s, off);
        if (threadIdx.x == 0) pooled[bc] = s * (1.0f / PIX_IN);
    }
}

// ---------------------------------------------------------------- routing ---
__global__ __launch_bounds__(128) void CEC_route_kernel(const float* __restrict__ pooled,
                                                        const float* __restrict__ rweight,
                                                        const float* __restrict__ rbias,
                                                        float* __restrict__ rw) {
    const int t = threadIdx.x;                    // 0..127  -> (b, e)
    const int b = t >> 2, e = t & 3;
    const float* pv = pooled + b * CIN;
    const float* wv = rweight + e * CIN;
    float s = rbias[e];
    #pragma unroll 4
    for (int c = 0; c < CIN; ++c) s += pv[c] * wv[c];
    rw[t] = 1.0f / (1.0f + __expf(-s));
}

// ----------------------------- per-sample bf16 weights, K-permuted layout ---
// wout[b][m][k'] with k' = (r*3+s)*128 + c   (source k = c*9 + tap)
__global__ __launch_bounds__(256) void CEC_wgen_kernel(const float* __restrict__ rw,
                                                       const float* __restrict__ ew,
                                                       bhalf* __restrict__ wout) {
    const int idx = blockIdx.x * 256 + threadIdx.x;   // < 32*294912
    const int b   = idx / WPS;
    const int rem = idx - b * WPS;                    // m*1152 + k'
    const int m   = rem / KDIM;
    const int kp  = rem - m * KDIM;                   // tap*128 + c
    const int tap = kp >> 7;
    const int c   = kp & 127;
    const int src = m * KDIM + c * 9 + tap;
    const float* r4 = rw + b * E_;
    float v = r4[0] * ew[src]
            + r4[1] * ew[WPS + src]
            + r4[2] * ew[2 * WPS + src]
            + r4[3] * ew[3 * WPS + src];
    wout[idx] = (bhalf)v;
}

// ------------------------------------------------------ implicit-GEMM conv --
union FragBF { uint4 u4[2]; bf16x16 v; };
union Pack8  { bhalf h[8]; uint4 q; };

__global__ __launch_bounds__(256) void CEC_conv_kernel(const float* __restrict__ x,
                                                       const bhalf* __restrict__ wbf,
                                                       float* __restrict__ out) {
    const int b  = blockIdx.y;
    const int n0 = blockIdx.x * 64;           // output-pixel tile

    __shared__ bhalf AT[3][256 * LDSTR];      // [m][k'] padded rows, 20 KB each
    __shared__ bhalf BT[3][64 * LDSTR];       // [n][k'] padded rows,  5 KB each

    const int t      = threadIdx.x;
    const int lane   = t & 31;
    const int wave   = t >> 5;
    const int lane16 = lane & 15;
    const int hi     = lane >> 4;
    const int waveM  = (wave & 3) * 64;       // 4 waves along M, 64 rows each
    const int waveN  = (wave >> 2) * 32;      // 2 waves along N, 32 cols each

    // B-tile (im2col) gather coords: thread -> (pixel nB, 8-channel group kbB)
    const int nB  = t & 63;
    const int kbB = (t >> 6) << 3;            // 0,8,16,24
    const int pix = n0 + nB;
    const int oh  = pix / HW_OUT;
    const int ow  = pix - oh * HW_OUT;
    const int ihb = oh * 2 - 1;
    const int iwb = ow * 2 - 1;

    const float* xb   = x + (size_t)b * CIN * PIX_IN;
    const bhalf* Wrow = wbf + (size_t)b * WPS + (size_t)t * KDIM;  // thread stages row m=t

    f32x8 acc[4][2] = {};
    float gf[8];
    bool  ginb = false;

    // A tile: pure bf16 global->LDS memcpy: 4x async b128 per thread.
    // IOFFSET applies to both LDS and global addresses (ISA 08 §4.4), so one
    // base pair + 4 immediate offsets covers the thread's 64-byte row slice.
    auto asyncA = [&](int step, bhalf* abuf) {
        const bhalf* src = Wrow + step * 32;
        AS3 bhalf* dst = (AS3 bhalf*)(abuf + t * LDSTR);
        asm volatile("global_load_async_to_lds_b128 %0, %1, off\n\t"
                     "global_load_async_to_lds_b128 %0, %1, off offset:16\n\t"
                     "global_load_async_to_lds_b128 %0, %1, off offset:32\n\t"
                     "global_load_async_to_lds_b128 %0, %1, off offset:48"
                     :: "v"(dst), "v"(src) : "memory");
    };

    // K-step -> single tap (r,s) + 32-channel slice: branchless, div-free gather
    auto gatherB = [&](int step) {
        const int tap = step >> 2;            // 0..8 (uniform)
        const int r   = tap / 3;              // uniform scalars
        const int s2  = tap - r * 3;
        const int c   = ((step & 3) << 5) + kbB;
        const int ih  = ihb + r;
        const int iw  = iwb + s2;
        ginb = ((unsigned)ih < (unsigned)HW_IN) & ((unsigned)iw < (unsigned)HW_IN);
        const int ihc = min(max(ih, 0), HW_IN - 1);
        const int iwc = min(max(iw, 0), HW_IN - 1);
        const float* p = xb + (size_t)c * PIX_IN + ihc * HW_IN + iwc;
        #pragma unroll
        for (int kk = 0; kk < 8; ++kk) gf[kk] = p[(size_t)kk * PIX_IN];
    };

    auto storeB = [&](bhalf* bbuf) {
        Pack8 v;
        #pragma unroll
        for (int kk = 0; kk < 8; ++kk) v.h[kk] = (bhalf)(ginb ? gf[kk] : 0.0f);
        *(uint4*)(bbuf + nB * LDSTR + kbB) = v.q;
    };

    // Fragments as explicit 16-byte LDS reads -> ds_load_b128.
    //   A (16x32 bf16, ISA 7.12.2): lane half 'hi' holds K {0..7,16..23} or
    //   {8..15,24..31}: bytes [hi*16, +16) and [hi*16+32, +16) of the row.
    //   B (32x16 bf16): same K split as A: lane half 'hi' holds K {0..7,16..23} or
    //   {8..15,24..31}: bytes [hi*16, +16) and [hi*16+32, +16) of the column-row.
    auto compute = [&](const bhalf* abuf, const bhalf* bbuf) {
        FragBF afr[4], bfr[2];
        #pragma unroll
        for (int i = 0; i < 4; ++i) {
            const uint4* q = (const uint4*)((const char*)(abuf + (waveM + i * 16 + lane16) * LDSTR)
                                            + hi * 16);
            afr[i].u4[0] = q[0];
            afr[i].u4[1] = q[2];          // +32 bytes
        }
        #pragma unroll
        for (int j = 0; j < 2; ++j) {
            const uint4* q = (const uint4*)((const char*)(bbuf + (waveN + j * 16 + lane16) * LDSTR)
                                            + hi * 16);
            bfr[j].u4[0] = q[0];
            bfr[j].u4[1] = q[2];          // +32 bytes
        }
        #pragma unroll
        for (int i = 0; i < 4; ++i)
            #pragma unroll
            for (int j = 0; j < 2; ++j)
                acc[i][j] = __builtin_amdgcn_wmma_f32_16x16x32_bf16(
                    false, afr[i].v, false, bfr[j].v,
                    (short)0, acc[i][j], false, false);
    };

    // ---- software pipeline: 3 LDS buffers, A async 2 steps ahead, B 1 ahead.
    // Each asyncA = 4 async instructions; async loads complete in order, so
    // s_wait_asynccnt 4 guarantees the older step's copy has landed while the
    // newest one is still in flight.
    bhalf *Ac = AT[0], *An = AT[1], *Ap = AT[2];
    bhalf *Bc = BT[0], *Bn = BT[1], *Bp = BT[2];

    asyncA(0, Ac);
    gatherB(0);
    storeB(Bc);
    asyncA(1, An);
    asm volatile("s_wait_asynccnt 0x4" ::: "memory");   // A(0) landed
    __syncthreads();

    for (int s = 0; s < KSTEPS; ++s) {
        if (s + 2 < KSTEPS) asyncA(s + 2, Ap);
        if (s + 1 < KSTEPS) gatherB(s + 1);
        compute(Ac, Bc);
        if (s + 1 < KSTEPS) storeB(Bn);
        if (s + 2 < KSTEPS) asm volatile("s_wait_asynccnt 0x4" ::: "memory");
        else                asm volatile("s_wait_asynccnt 0x0" ::: "memory");
        __syncthreads();
        bhalf* ta = Ac; Ac = An; An = Ap; Ap = ta;
        bhalf* tb = Bc; Bc = Bn; Bn = Bp; Bp = tb;
    }

    // ---- store: C/D layout — VGPR r: lanes 0-15 M=r, lanes 16-31 M=r+8 ----
    #pragma unroll
    for (int i = 0; i < 4; ++i) {
        const int mbase = waveM + i * 16 + hi * 8;
        #pragma unroll
        for (int j = 0; j < 2; ++j) {
            const int n = n0 + waveN + j * 16 + lane16;
            float* op = out + ((size_t)b * COUT + mbase) * PIX_OUT + n;
            #pragma unroll
            for (int r = 0; r < 8; ++r)
                op[(size_t)r * PIX_OUT] = acc[i][j][r];
        }
    }
}

// ------------------------------------------------------------------ launch --
extern "C" void kernel_launch(void* const* d_in, const int* in_sizes, int n_in,
                              void* d_out, int out_size, void* d_ws, size_t ws_size,
                              hipStream_t stream) {
    const float* x    = (const float*)d_in[0];
    const float* rwgt = (const float*)d_in[1];
    const float* rbia = (const float*)d_in[2];
    const float* ewgt = (const float*)d_in[3];
    float* out = (float*)d_out;

    char* ws = (char*)d_ws;
    float* pooled = (float*)ws;                   // 32*128 f32 = 16 KB
    float* rw     = (float*)(ws + 16384);         // 32*4 f32
    bhalf* wbf    = (bhalf*)(ws + 32768);         // 32*294912 bf16 ≈ 18.9 MB

    CEC_pool_kernel <<<B_ * CIN, 256, 0, stream>>>(x, pooled);
    CEC_route_kernel<<<1, 128, 0, stream>>>(pooled, rwgt, rbia, rw);
    CEC_wgen_kernel <<<(B_ * WPS) / 256, 256, 0, stream>>>(rw, ewgt, wbf);

    dim3 grid(PIX_OUT / 64, B_);                  // 49 x 32
    CEC_conv_kernel <<<grid, 256, 0, stream>>>(x, wbf, out);
}
